// MultiHeadSelfAttention_60808146977167
// MI455X (gfx1250) — matrix operand order repaired and checked by reading, not compiled, not run on hardware
//
#include <hip/hip_runtime.h>
#include <type_traits>

typedef __attribute__((ext_vector_type(16))) __bf16 v16bf;
typedef __attribute__((ext_vector_type(8)))  __bf16 v8bf;
typedef __attribute__((ext_vector_type(4)))  __bf16 v4bf;
typedef __attribute__((ext_vector_type(8)))  float  v8f;

#define D_MODEL 1024
#define N_HEADS 16
#define HEAD_DIM 64
#define SEQ_T   2048
#define BATCH   2
#define M_ROWS  (BATCH * SEQ_T)   // 4096

// ---- WMMA fragment layouts (CDNA5 ISA, 16-bit operands) ----
// A 16x32 (MxK): lane L (m = L&15, half = L>>4):
//   elements 0..7  <- k = 8*half + 0..7        (contiguous)
//   elements 8..15 <- k = 16 + 8*half + 0..7   (contiguous)
// B 32x16 (KxN): lane L (n = L&15, half = L>>4): same k order as A
// C/D 16x16 f32: VGPR r: m = r + 8*half, n = L&15

__device__ __forceinline__ v8bf ld8(const __bf16* p) { return *(const v8bf*)p; }

__device__ __forceinline__ v16bf frag_a(const __bf16* row, int ko, int half) {
  v8bf lo = ld8(row + ko + 8 * half);
  v8bf hi = ld8(row + ko + 16 + 8 * half);
  return __builtin_shufflevector(lo, hi, 0,1,2,3,4,5,6,7,8,9,10,11,12,13,14,15);
}
__device__ __forceinline__ v16bf frag_b(const __bf16* row, int ko, int half) {
  v8bf lo = ld8(row + ko + 8 * half);
  v8bf hi = ld8(row + ko + 16 + 8 * half);
  return __builtin_shufflevector(lo, hi, 0,1,2,3,4,5,6,7,8,9,10,11,12,13,14,15);
}

__device__ __forceinline__ v8f wmma_bf16(v16bf a, v16bf b, v8f c) {
  return __builtin_amdgcn_wmma_f32_16x16x32_bf16(
      false, a, false, b, (short)0, c, false, false);
}

// DPP rotate within each 16-lane row (wave32: two independent rows)
template <int CTRL>
__device__ __forceinline__ float dpp_rot(float x) {
  int r = __builtin_amdgcn_update_dpp(0, __builtin_bit_cast(int, x),
                                      CTRL, 0xf, 0xf, true);
  return __builtin_bit_cast(float, r);
}
// max across the 16 lanes of a DPP row (row_ror 1,2,4,8 butterfly)
__device__ __forceinline__ float rowmax16(float x) {
  x = fmaxf(x, dpp_rot<0x121>(x));
  x = fmaxf(x, dpp_rot<0x122>(x));
  x = fmaxf(x, dpp_rot<0x124>(x));
  x = fmaxf(x, dpp_rot<0x128>(x));
  return x;
}

// ============================================================================
// Kernel 0: f32 -> bf16 bulk convert with scale (4 elems/thread)
// ============================================================================
__global__ void k_cvt_bf16(const float* __restrict__ in,
                           __bf16* __restrict__ out, int n4, float scale) {
  int i = blockIdx.x * blockDim.x + threadIdx.x;
  if (i < n4) {
    float4 f = ((const float4*)in)[i];
    v4bf o;
    o[0] = (__bf16)(f.x * scale); o[1] = (__bf16)(f.y * scale);
    o[2] = (__bf16)(f.z * scale); o[3] = (__bf16)(f.w * scale);
    ((v4bf*)out)[i] = o;
  }
}

// ============================================================================
// Kernel 1: Y = Xb @ Wb^T + bias*bscale (all bf16), templated output layout.
//   MODE 0: scatter to [B,H,T,Dh]  (Q and K)
//   MODE 1: packed store to [B,H,Dh,T] (V transposed; 8 consecutive t / lane)
// block = 128 (4 waves). Wave -> 16x64 output strip. grid = (256, 4)
// ============================================================================
template <int MODE>
__global__ void k_proj(const __bf16* __restrict__ Xb,
                       const __bf16* __restrict__ Wb,
                       const float* __restrict__ Bias,
                       __bf16* __restrict__ Out, float bscale) {
  const int lane = threadIdx.x & 31;
  const int wave = threadIdx.x >> 5;
  const int mt = blockIdx.x;
  const int nt0 = (blockIdx.y * 4 + wave) * 4;
  const int half = lane >> 4;
  const int mrow = lane & 15;
  const int ncol = lane & 15;

  const __bf16* xrow = Xb + (size_t)(mt * 16 + mrow) * D_MODEL;
  const __bf16* wrow0 = Wb + (size_t)((nt0 + 0) * 16 + ncol) * D_MODEL;
  const __bf16* wrow1 = Wb + (size_t)((nt0 + 1) * 16 + ncol) * D_MODEL;
  const __bf16* wrow2 = Wb + (size_t)((nt0 + 2) * 16 + ncol) * D_MODEL;
  const __bf16* wrow3 = Wb + (size_t)((nt0 + 3) * 16 + ncol) * D_MODEL;

  v8f acc0 = {}, acc1 = {}, acc2 = {}, acc3 = {};
  for (int ko = 0; ko < D_MODEL; ko += 32) {
    v16bf a = frag_a(xrow, ko, half);
    acc0 = wmma_bf16(a, frag_b(wrow0, ko, half), acc0);
    acc1 = wmma_bf16(a, frag_b(wrow1, ko, half), acc1);
    acc2 = wmma_bf16(a, frag_b(wrow2, ko, half), acc2);
    acc3 = wmma_bf16(a, frag_b(wrow3, ko, half), acc3);
  }

  v8f* accs[4] = {&acc0, &acc1, &acc2, &acc3};
#pragma unroll
  for (int j = 0; j < 4; ++j) {
    const int n = (nt0 + j) * 16 + ncol;
    const int h = n >> 6, dh = n & 63;
    const float bval = Bias[n] * bscale;
    if (MODE == 0) {
#pragma unroll
      for (int r = 0; r < 8; ++r) {
        int m = mt * 16 + r + 8 * half;
        int bb = m >> 11, t = m & 2047;
        Out[(((size_t)bb * N_HEADS + h) * SEQ_T + t) * HEAD_DIM + dh] =
            (__bf16)((*accs[j])[r] + bval);
      }
    } else {
      v8bf pack;
#pragma unroll
      for (int r = 0; r < 8; ++r) pack[r] = (__bf16)((*accs[j])[r] + bval);
      const int m0 = mt * 16 + 8 * half;
      const int bb = m0 >> 11, t0 = m0 & 2047;
      *(v8bf*)&Out[(((size_t)bb * N_HEADS + h) * HEAD_DIM + dh) * SEQ_T + t0] =
          pack;
    }
  }
}

// ============================================================================
// Kernel 2: causal flash attention. One wave = one 16-query tile of one (b,h).
// Q (pre-scaled by 1/sqrt(Dh)),K: bf16 [B,H,T,Dh]; VT: bf16 [B,H,Dh,T].
// Output: bf16 [B,T,H*Dh]. block = 128 (4 waves); grid = (32, H, B).
// Exactly one partial (masked) 32-key block per query tile; all earlier
// blocks run the maskless fast path. 9 WMMAs per block.
// ============================================================================
__global__ void k_attn(const __bf16* __restrict__ Q,
                       const __bf16* __restrict__ K,
                       const __bf16* __restrict__ VT,
                       __bf16* __restrict__ O) {
  __shared__ __bf16 pbuf[4][16 * 32];   // per-wave P tile (16 q x 32 keys)

  const int lane = threadIdx.x & 31;
  const int wave = threadIdx.x >> 5;
  const int qt = blockIdx.x * 4 + wave;
  const int h = blockIdx.y;
  const int b = blockIdx.z;
  const int half = lane >> 4;
  const int mrow = lane & 15;
  const int ncol = lane & 15;

  const size_t bh = (size_t)b * N_HEADS + h;
  const size_t bh_row0 = bh * SEQ_T;

  const __bf16* qrow = Q + (bh_row0 + qt * 16 + mrow) * HEAD_DIM;
  v16bf aq0 = frag_a(qrow, 0, half);
  v16bf aq1 = frag_a(qrow, 32, half);

  const __bf16* vt0 = VT + (bh * HEAD_DIM + 0 * 16 + ncol) * SEQ_T;
  const __bf16* vt1 = VT + (bh * HEAD_DIM + 1 * 16 + ncol) * SEQ_T;
  const __bf16* vt2 = VT + (bh * HEAD_DIM + 2 * 16 + ncol) * SEQ_T;
  const __bf16* vt3 = VT + (bh * HEAD_DIM + 3 * 16 + ncol) * SEQ_T;

  v16bf ones;
#pragma unroll
  for (int e = 0; e < 16; ++e) ones[e] = (__bf16)1.0f;

  v8f o0 = {}, o1 = {}, o2 = {}, o3 = {};
  float mi[8], li[8], alph[8];
#pragma unroll
  for (int r = 0; r < 8; ++r) { mi[r] = -__builtin_inff(); li[r] = 0.f; }

  auto body = [&](int keybase, auto masked_t) {
    constexpr bool MASKED = decltype(masked_t)::value;

    // ---- scores S[16x32] = Q x K^T (scale pre-folded into Q) ----
    v8f s0 = {}, s1 = {};
    {
      const __bf16* krow0 = K + (bh_row0 + keybase + ncol) * HEAD_DIM;
      const __bf16* krow1 = K + (bh_row0 + keybase + 16 + ncol) * HEAD_DIM;
      s0 = wmma_bf16(aq0, frag_b(krow0, 0, half), s0);
      s0 = wmma_bf16(aq1, frag_b(krow0, 32, half), s0);
      s1 = wmma_bf16(aq0, frag_b(krow1, 0, half), s1);
      s1 = wmma_bf16(aq1, frag_b(krow1, 32, half), s1);
    }

    // ---- (mask +) running max via DPP + exp; P -> LDS ----
#pragma unroll
    for (int r = 0; r < 8; ++r) {
      float v0 = s0[r];
      float v1 = s1[r];
      if (MASKED) {
        const int q = qt * 16 + r + 8 * half;
        v0 = (keybase + ncol <= q)      ? v0 : -__builtin_inff();
        v1 = (keybase + 16 + ncol <= q) ? v1 : -__builtin_inff();
      }
      float rm = rowmax16(fmaxf(v0, v1));
      float mn = fmaxf(mi[r], rm);
      float a  = __expf(mi[r] - mn);
      float p0 = __expf(v0 - mn);
      float p1 = __expf(v1 - mn);

      mi[r] = mn;
      alph[r] = a;
      o0[r] = o0[r] * a;
      o1[r] = o1[r] * a;
      o2[r] = o2[r] * a;
      o3[r] = o3[r] * a;

      const int m = r + 8 * half;
      pbuf[wave][m * 32 + ncol]      = (__bf16)p0;
      pbuf[wave][m * 32 + 16 + ncol] = (__bf16)p1;
    }
    __builtin_amdgcn_wave_barrier();
    asm volatile("s_wait_dscnt 0" ::: "memory");   // C->A relayout through LDS

    // P as A-fragment: two ds_load_b128
    v16bf ap;
    {
      const __bf16* prow = &pbuf[wave][mrow * 32];
      v8bf lo = *(const v8bf*)(prow + 8 * half);
      v8bf hi = *(const v8bf*)(prow + 16 + 8 * half);
      ap = __builtin_shufflevector(lo, hi,
                                   0,1,2,3,4,5,6,7,8,9,10,11,12,13,14,15);
    }
    __builtin_amdgcn_wave_barrier();

    // ---- row sums via WMMA: P @ ones -> every column = rowsum ----
    {
      v8f z = {};
      v8f ps = wmma_bf16(ap, ones, z);
#pragma unroll
      for (int r = 0; r < 8; ++r) li[r] = alph[r] * li[r] + ps[r];
    }

    // ---- PV: O += P(16x32) x V(32x64) ----
    o0 = wmma_bf16(ap, frag_b(vt0, keybase, half), o0);
    o1 = wmma_bf16(ap, frag_b(vt1, keybase, half), o1);
    o2 = wmma_bf16(ap, frag_b(vt2, keybase, half), o2);
    o3 = wmma_bf16(ap, frag_b(vt3, keybase, half), o3);
  };

  // Blocks [0, nfull) are entirely below the diagonal: maskless fast path.
  // Block nfull is the single diagonal (partial) block.
  const int nfull = (qt * 16) / 32;
  for (int blk = 0; blk < nfull; ++blk)
    body(blk * 32, std::integral_constant<bool, false>{});
  body(nfull * 32, std::integral_constant<bool, true>{});

  // ---- epilogue: normalize, store bf16 [B,T,H*Dh] ----
#pragma unroll
  for (int r = 0; r < 8; ++r) {
    const int q = qt * 16 + r + 8 * half;
    const float inv = 1.f / li[r];
    const size_t base = ((size_t)b * SEQ_T + q) * D_MODEL + h * HEAD_DIM;
    O[base + ncol]      = (__bf16)(o0[r] * inv);
    O[base + 16 + ncol] = (__bf16)(o1[r] * inv);
    O[base + 32 + ncol] = (__bf16)(o2[r] * inv);
    O[base + 48 + ncol] = (__bf16)(o3[r] * inv);
  }
}

// ============================================================================
// Kernel 3: out = A(bf16) @ Wo^T(bf16) + bo, f32 out [4096,1024].
// ============================================================================
__global__ void k_out_proj(const __bf16* __restrict__ A,
                           const __bf16* __restrict__ Wb,
                           const float* __restrict__ Bias,
                           float* __restrict__ Out) {
  const int lane = threadIdx.x & 31;
  const int wave = threadIdx.x >> 5;
  const int mt = blockIdx.x;
  const int nt0 = (blockIdx.y * 4 + wave) * 4;
  const int half = lane >> 4;
  const int mrow = lane & 15;
  const int ncol = lane & 15;

  const __bf16* arow = A + (size_t)(mt * 16 + mrow) * D_MODEL;
  const __bf16* wrow0 = Wb + (size_t)((nt0 + 0) * 16 + ncol) * D_MODEL;
  const __bf16* wrow1 = Wb + (size_t)((nt0 + 1) * 16 + ncol) * D_MODEL;
  const __bf16* wrow2 = Wb + (size_t)((nt0 + 2) * 16 + ncol) * D_MODEL;
  const __bf16* wrow3 = Wb + (size_t)((nt0 + 3) * 16 + ncol) * D_MODEL;

  v8f acc0 = {}, acc1 = {}, acc2 = {}, acc3 = {};
  for (int ko = 0; ko < D_MODEL; ko += 32) {
    v16bf a = frag_a(arow, ko, half);
    acc0 = wmma_bf16(a, frag_b(wrow0, ko, half), acc0);
    acc1 = wmma_bf16(a, frag_b(wrow1, ko, half), acc1);
    acc2 = wmma_bf16(a, frag_b(wrow2, ko, half), acc2);
    acc3 = wmma_bf16(a, frag_b(wrow3, ko, half), acc3);
  }

  v8f* accs[4] = {&acc0, &acc1, &acc2, &acc3};
#pragma unroll
  for (int j = 0; j < 4; ++j) {
    const int n = (nt0 + j) * 16 + ncol;
    const float bval = Bias[n];
#pragma unroll
    for (int r = 0; r < 8; ++r) {
      int m = mt * 16 + r + 8 * half;
      Out[(size_t)m * D_MODEL + n] = (*accs[j])[r] + bval;
    }
  }
}

// ============================================================================
extern "C" void kernel_launch(void* const* d_in, const int* in_sizes, int n_in,
                              void* d_out, int out_size, void* d_ws, size_t ws_size,
                              hipStream_t stream) {
  (void)in_sizes; (void)n_in; (void)out_size; (void)ws_size;
  const float* x  = (const float*)d_in[0];
  const float* wq = (const float*)d_in[1];
  const float* bq = (const float*)d_in[2];
  const float* wk = (const float*)d_in[3];
  const float* bk = (const float*)d_in[4];
  const float* wv = (const float*)d_in[5];
  const float* bv = (const float*)d_in[6];
  const float* wo = (const float*)d_in[7];
  const float* bo = (const float*)d_in[8];

  const size_t xe = (size_t)M_ROWS * D_MODEL;     // 4 Mi elems
  const size_t we = (size_t)D_MODEL * D_MODEL;    // 1 Mi elems
  __bf16* xb  = (__bf16*)d_ws;
  __bf16* wqb = xb  + xe;
  __bf16* wkb = wqb + we;
  __bf16* wvb = wkb + we;
  __bf16* wob = wvb + we;
  __bf16* qb  = wob + we;
  __bf16* kb  = qb + xe;
  __bf16* vtb = kb + xe;                          // [B,H,Dh,T]
  __bf16* ab  = vtb + xe;                         // total 48 MB

  const float qscale = 0.125f;   // 1/sqrt(HEAD_DIM), folded into W_q and b_q
  {
    int n4x = (int)(xe / 4), n4w = (int)(we / 4);
    dim3 cb(256);
    k_cvt_bf16<<<dim3((n4x + 255) / 256), cb, 0, stream>>>(x,  xb,  n4x, 1.f);
    k_cvt_bf16<<<dim3((n4w + 255) / 256), cb, 0, stream>>>(wq, wqb, n4w, qscale);
    k_cvt_bf16<<<dim3((n4w + 255) / 256), cb, 0, stream>>>(wk, wkb, n4w, 1.f);
    k_cvt_bf16<<<dim3((n4w + 255) / 256), cb, 0, stream>>>(wv, wvb, n4w, 1.f);
    k_cvt_bf16<<<dim3((n4w + 255) / 256), cb, 0, stream>>>(wo, wob, n4w, 1.f);
  }

  dim3 blk(128);
  dim3 gemm_grid(M_ROWS / 16, D_MODEL / 16 / 16);   // (256, 4)
  k_proj<0><<<gemm_grid, blk, 0, stream>>>(xb, wqb, bq, qb, qscale);
  k_proj<0><<<gemm_grid, blk, 0, stream>>>(xb, wkb, bk, kb, 1.f);
  k_proj<1><<<gemm_grid, blk, 0, stream>>>(xb, wvb, bv, vtb, 1.f);

  dim3 attn_grid(SEQ_T / 16 / 4, N_HEADS, BATCH);   // (32, 16, 2)
  k_attn<<<attn_grid, blk, 0, stream>>>(qb, kb, vtb, ab);

  k_out_proj<<<gemm_grid, blk, 0, stream>>>(ab, wob, bo, (float*)d_out);
}
